// SelfAttentionBlock2d_85873576117032
// MI455X (gfx1250) — matrix operand order repaired and checked by reading, not compiled, not run on hardware
//
#include <hip/hip_runtime.h>
#include <hip/hip_bf16.h>

#define N_B   4
#define C_DIM 256
#define S_DIM 4096
#define E_DIM 128

typedef __attribute__((ext_vector_type(16))) _Float16 v16h;
typedef __attribute__((ext_vector_type(8)))  _Float16 v8h;
typedef __attribute__((ext_vector_type(8)))  float    v8f;

#if __has_builtin(__builtin_amdgcn_tensor_load_to_lds)
#define USE_TDM 1
typedef unsigned int tdm_u32x4 __attribute__((ext_vector_type(4)));
typedef int          tdm_i32x4 __attribute__((ext_vector_type(4)));
typedef int          tdm_i32x8 __attribute__((ext_vector_type(8)));
#endif

__device__ __forceinline__ v16h join16(v8h lo, v8h hi) {
    v16h r;
#pragma unroll
    for (int i = 0; i < 8; i++) { r[i] = lo[i]; r[i + 8] = hi[i]; }
    return r;
}

#ifdef USE_TDM
// Issue one 2D-tile TDM load: global (dim0 x dim1, row stride stride0 elements,
// 2-byte elements) -> LDS at lds_byte_off. D# fields per CDNA5 ISA ch.8.
__device__ __forceinline__ void tdm_load_2d(unsigned lds_byte_off, const void* gptr,
                                            unsigned dim0, unsigned dim1,
                                            unsigned stride0) {
    const unsigned long long ga = (unsigned long long)(uintptr_t)gptr;
    tdm_u32x4 g0;
    g0[0] = 1u;                                            // count=1 (user descriptor)
    g0[1] = lds_byte_off;                                  // lds_addr
    g0[2] = (unsigned)ga;                                  // global_addr[31:0]
    g0[3] = (unsigned)((ga >> 32) & 0x01FFFFFFu) | (2u << 30); // addr[56:32], type=2
    tdm_i32x8 g1;
    g1[0] = (int)(1u << 16);                               // data_size=1 (2B), mask=0
    g1[1] = (int)((dim0 & 0xFFFFu) << 16);                 // tensor_dim0 lo16 @bit48
    g1[2] = (int)((dim0 >> 16) | ((dim1 & 0xFFFFu) << 16));// tensor_dim0 hi / dim1 lo
    g1[3] = (int)((dim1 >> 16) | ((dim0 & 0xFFFFu) << 16));// dim1 hi / tile_dim0
    g1[4] = (int)(dim1 & 0xFFFFu);                         // tile_dim1, tile_dim2=0
    g1[5] = (int)stride0;                                  // tensor_dim0_stride lo32
    g1[6] = 0;                                             // stride0 hi / stride1 lo
    g1[7] = 0;
    tdm_i32x4 z4 = {0, 0, 0, 0};
#if __clang_major__ >= 23
    tdm_i32x8 z8 = {0, 0, 0, 0, 0, 0, 0, 0};
    __builtin_amdgcn_tensor_load_to_lds(g0, g1, z4, z4, z8, 0);
#else
    __builtin_amdgcn_tensor_load_to_lds(g0, g1, z4, z4, 0);
#endif
}

__device__ __forceinline__ void tdm_wait_all() {
#if __has_builtin(__builtin_amdgcn_s_wait_tensorcnt)
    __builtin_amdgcn_s_wait_tensorcnt(0);
#else
    asm volatile("s_wait_tensorcnt 0x0" ::: "memory");
#endif
}
#endif

// ---------------------------------------------------------------------------
// Kernel 1: fused Q/K/V projection (1x1 convs). x:[N,C,S] f32 -> q,k:[N,S,E] f16,
// v stored transposed as [N,E,S] f16 so PV B-fragments are contiguous in t.
// ---------------------------------------------------------------------------
__global__ __launch_bounds__(256) void proj_kernel(
    const float* __restrict__ x,
    const float* __restrict__ Wq, const float* __restrict__ bq,
    const float* __restrict__ Wk, const float* __restrict__ bk,
    const float* __restrict__ Wv, const float* __restrict__ bv,
    _Float16* __restrict__ qf, _Float16* __restrict__ kf,
    _Float16* __restrict__ vT)
{
    const int s  = blockIdx.x * 256 + threadIdx.x;
    const int e0 = blockIdx.y * 8;
    const int n  = blockIdx.z;

    float aq[8], ak[8], av[8];
#pragma unroll
    for (int j = 0; j < 8; j++) { aq[j] = bq[e0 + j]; ak[j] = bk[e0 + j]; av[j] = bv[e0 + j]; }

    const float* xp = x + (size_t)n * C_DIM * S_DIM + s;
    for (int c = 0; c < C_DIM; c++) {
        const float xv = xp[(size_t)c * S_DIM];          // coalesced across threads
#pragma unroll
        for (int j = 0; j < 8; j++) {                    // uniform weights -> scalar loads
            aq[j] = fmaf(Wq[(e0 + j) * C_DIM + c], xv, aq[j]);
            ak[j] = fmaf(Wk[(e0 + j) * C_DIM + c], xv, ak[j]);
            av[j] = fmaf(Wv[(e0 + j) * C_DIM + c], xv, av[j]);
        }
    }

    const size_t se = ((size_t)(n * S_DIM + s)) * E_DIM + e0;
#pragma unroll
    for (int j = 0; j < 8; j++) {
        qf[se + j] = (_Float16)aq[j];
        kf[se + j] = (_Float16)ak[j];
        vT[((size_t)(n * E_DIM + e0 + j)) * S_DIM + s] = (_Float16)av[j];
    }
}

// ---------------------------------------------------------------------------
// Kernel 2: flash attention, workgroup-cooperative + TDM double buffering.
//   block = 256 threads = 8 waves = 128 query rows sharing each 64-key slab.
//   Slab i+1 is DMA'd (tensor_load_to_lds, TENSORcnt) into the spare LDS
//   buffer while all waves run WMMAs on slab i; one barrier per iteration.
// ---------------------------------------------------------------------------
__global__ __launch_bounds__(256) void attn_kernel(
    const _Float16* __restrict__ qf, const _Float16* __restrict__ kf,
    const _Float16* __restrict__ vT, _Float16* __restrict__ ao)
{
    __shared__ __align__(32) _Float16 sK[2][64 * 128];   // [t_local][e]   2x16 KB
    __shared__ __align__(32) _Float16 sV[2][128 * 64];   // [e][t_local]   2x16 KB
    __shared__ __align__(32) _Float16 sP[8][16 * 64];    // per-wave P       16 KB

    const int tid  = threadIdx.x;
    const int lane = tid & 31;
    const int w    = tid >> 5;
    const int n    = blockIdx.x >> 5;                    // 32 blocks per batch
    const int qb   = (blockIdx.x & 31) * 128;            // block's first query row
    const int q0   = qb + w * 16;                        // wave's query tile
    const int hsel = (lane < 16) ? 0 : 1;
    const int ln   = lane & 15;

    // ---- preload Q A-fragments (K=128 -> 4 fragments of 16x32) ----
    v16h qa[4];
    const _Float16* qrow = qf + ((size_t)(n * S_DIM + q0 + ln)) * E_DIM;
#pragma unroll
    for (int kk = 0; kk < 4; kk++) {
        const int base = kk * 32 + hsel * 8;
        v8h lo = *(const v8h*)(qrow + base);
        v8h hi = *(const v8h*)(qrow + base + 16);
        qa[kk] = join16(lo, hi);
    }

    v8f o[8];
    float m[8], l[8];
#pragma unroll
    for (int i = 0; i < 8; i++) { v8f z = {}; o[i] = z; m[i] = -3.0e38f; l[i] = 0.f; }

    // ---- stage slab 0 into buffer 0 ----
#ifdef USE_TDM
    if (w == 0)
        tdm_load_2d((unsigned)(uintptr_t)&sK[0][0], kf + ((size_t)n * S_DIM) * E_DIM,
                    E_DIM, 64, E_DIM);
    else if (w == 1)
        tdm_load_2d((unsigned)(uintptr_t)&sV[0][0], vT + (size_t)n * E_DIM * S_DIM,
                    64, E_DIM, S_DIM);
#else
    {
        const v8h* srcK = (const v8h*)(kf + ((size_t)n * S_DIM) * E_DIM);
#pragma unroll
        for (int j = 0; j < 4; j++) ((v8h*)sK[0])[tid + j * 256] = srcK[tid + j * 256];
#pragma unroll
        for (int j = 0; j < 4; j++) {
            const int c = tid + j * 256;
            ((v8h*)sV[0])[c] = *(const v8h*)(vT + ((size_t)(n * E_DIM + (c >> 3))) * S_DIM
                                             + (c & 7) * 8);
        }
    }
#endif

    for (int t0 = 0; t0 < S_DIM; t0 += 64) {
        const int b = (t0 >> 6) & 1;
#ifdef USE_TDM
        tdm_wait_all();                                   // slab b DMA complete
#endif
        __syncthreads();                                  // publish slab b; spare buf free

        // ---- kick off staging of the NEXT slab into the spare buffer ----
        if (t0 + 64 < S_DIM) {
            const int nb = b ^ 1;
            const int t1 = t0 + 64;
#ifdef USE_TDM
            if (w == 0)
                tdm_load_2d((unsigned)(uintptr_t)&sK[nb][0],
                            kf + ((size_t)n * S_DIM + t1) * E_DIM, E_DIM, 64, E_DIM);
            else if (w == 1)
                tdm_load_2d((unsigned)(uintptr_t)&sV[nb][0],
                            vT + (size_t)n * E_DIM * S_DIM + t1, 64, E_DIM, S_DIM);
#else
            __builtin_prefetch(kf + ((size_t)n * S_DIM + t1) * E_DIM + tid * 32, 0, 1);
            const v8h* srcK = (const v8h*)(kf + ((size_t)n * S_DIM + t1) * E_DIM);
#pragma unroll
            for (int j = 0; j < 4; j++) ((v8h*)sK[nb])[tid + j * 256] = srcK[tid + j * 256];
#pragma unroll
            for (int j = 0; j < 4; j++) {
                const int c = tid + j * 256;
                ((v8h*)sV[nb])[c] = *(const v8h*)(vT + ((size_t)(n * E_DIM + (c >> 3))) * S_DIM
                                                  + t1 + (c & 7) * 8);
            }
#endif
        }

        const _Float16* Kb = sK[b];
        const _Float16* Vb = sV[b];

        // ---- scores: four 16x16 tiles over the 64-key slab ----
        v8f sc[4];
#pragma unroll
        for (int j = 0; j < 4; j++) {
            v8f c = {};
            const _Float16* krow = Kb + (j * 16 + ln) * 128;
#pragma unroll
            for (int kk = 0; kk < 4; kk++) {
                v16h kb = join16(*(const v8h*)(krow + kk * 32 + hsel * 8),
                                 *(const v8h*)(krow + kk * 32 + hsel * 8 + 16)); // two 16B chunks
                c = __builtin_amdgcn_wmma_f32_16x16x32_f16(
                        false, qa[kk], false, kb, (short)0, c, false, false);
            }
            sc[j] = c;
        }

        // ---- online softmax over 64 keys (amortized reductions) ----
        float alpha[8];
#pragma unroll
        for (int r = 0; r < 8; r++) {
            float mx = fmaxf(fmaxf(sc[0][r], sc[1][r]), fmaxf(sc[2][r], sc[3][r]));
            mx = fmaxf(mx, __shfl_xor(mx, 1, 32));
            mx = fmaxf(mx, __shfl_xor(mx, 2, 32));
            mx = fmaxf(mx, __shfl_xor(mx, 4, 32));
            mx = fmaxf(mx, __shfl_xor(mx, 8, 32));
            const float mn = fmaxf(m[r], mx);
            alpha[r] = __expf(m[r] - mn);
            m[r] = mn;
        }
#pragma unroll
        for (int r = 0; r < 8; r++) {
            const int row = r + hsel * 8;                 // C-layout row this lane owns
            float sum = 0.f;
#pragma unroll
            for (int j = 0; j < 4; j++) {
                const float p = __expf(sc[j][r] - m[r]);
                sP[w][row * 64 + j * 16 + ln] = (_Float16)p;
                sum += p;
            }
            sum += __shfl_xor(sum, 1, 32);
            sum += __shfl_xor(sum, 2, 32);
            sum += __shfl_xor(sum, 4, 32);
            sum += __shfl_xor(sum, 8, 32);
            l[r] = l[r] * alpha[r] + sum;
        }
#pragma unroll
        for (int et = 0; et < 8; et++)
#pragma unroll
            for (int r = 0; r < 8; r++)
                o[et][r] *= alpha[r];

        // ---- O += P x V : 2 K=32 steps x 8 e-tiles ----
#pragma unroll
        for (int kk2 = 0; kk2 < 2; kk2++) {
            const int pb = kk2 * 32 + hsel * 8;           // A-layout chunk base
            v8h plo = *(const v8h*)&sP[w][ln * 64 + pb];
            v8h phi = *(const v8h*)&sP[w][ln * 64 + pb + 16];
            const v16h pf = join16(plo, phi);
#pragma unroll
            for (int et = 0; et < 8; et++) {
                v16h vb = join16(*(const v8h*)(Vb + (et * 16 + ln) * 64 + pb),
                                 *(const v8h*)(Vb + (et * 16 + ln) * 64 + pb + 16));
                o[et] = __builtin_amdgcn_wmma_f32_16x16x32_f16(
                            false, pf, false, vb, (short)0, o[et], false, false);
            }
        }
    }

    // ---- normalize & store attention output [N,S,E] f16 ----
    float inv[8];
#pragma unroll
    for (int r = 0; r < 8; r++) inv[r] = (l[r] > 0.f) ? (1.f / l[r]) : 0.f;
#pragma unroll
    for (int et = 0; et < 8; et++) {
#pragma unroll
        for (int r = 0; r < 8; r++) {
            const int s = q0 + r + hsel * 8;
            const int e = et * 16 + ln;
            ao[((size_t)(n * S_DIM + s)) * E_DIM + e] = (_Float16)(o[et][r] * inv[r]);
        }
    }
}

// ---------------------------------------------------------------------------
// Kernel 3: y = Wo * out + bo, residual add. Attention row cached in registers
// and reused across 16 output channels per block.
// ---------------------------------------------------------------------------
__global__ __launch_bounds__(256) void outproj_kernel(
    const float* __restrict__ x, const float* __restrict__ Wo,
    const float* __restrict__ bo, const _Float16* __restrict__ ao,
    float* __restrict__ out)
{
    const int s  = blockIdx.x * 256 + threadIdx.x;
    const int c0 = blockIdx.y * 16;
    const int n  = blockIdx.z;

    v8h a[16];
    const _Float16* ap = ao + ((size_t)(n * S_DIM + s)) * E_DIM;
#pragma unroll
    for (int i = 0; i < 16; i++) a[i] = *(const v8h*)(ap + i * 8);

    for (int j = 0; j < 16; j++) {
        const int c = c0 + j;
        float acc = bo[c];
        const float* wrow = Wo + (size_t)c * E_DIM;       // uniform -> scalar loads
#pragma unroll
        for (int i = 0; i < 16; i++)
#pragma unroll
            for (int u = 0; u < 8; u++)
                acc = fmaf((float)a[i][u], wrow[i * 8 + u], acc);
        const size_t idx = ((size_t)(n * C_DIM + c)) * S_DIM + s;
        out[idx] = x[idx] + acc;
    }
}

extern "C" void kernel_launch(void* const* d_in, const int* in_sizes, int n_in,
                              void* d_out, int out_size, void* d_ws, size_t ws_size,
                              hipStream_t stream) {
    const float* x  = (const float*)d_in[0];
    const float* Wq = (const float*)d_in[1];
    const float* bq = (const float*)d_in[2];
    const float* Wk = (const float*)d_in[3];
    const float* bk = (const float*)d_in[4];
    const float* Wv = (const float*)d_in[5];
    const float* bv = (const float*)d_in[6];
    const float* Wo = (const float*)d_in[7];
    const float* bo = (const float*)d_in[8];
    float* out = (float*)d_out;

    const size_t QSZ = (size_t)N_B * S_DIM * E_DIM;       // 2,097,152 f16 each
    _Float16* qf = (_Float16*)d_ws;                       // [N,S,E]
    _Float16* kf = qf + QSZ;                              // [N,S,E]
    _Float16* vT = kf + QSZ;                              // [N,E,S]
    _Float16* ao = vT + QSZ;                              // [N,S,E]  (16 MB total)

    dim3 g1(S_DIM / 256, E_DIM / 8, N_B);
    proj_kernel<<<g1, 256, 0, stream>>>(x, Wq, bq, Wk, bk, Wv, bv, qf, kf, vT);

    // 8 waves per block, 128 query rows per block -> 128 blocks
    attn_kernel<<<dim3(N_B * (S_DIM / 128)), 256, 0, stream>>>(qf, kf, vT, ao);

    dim3 g3(S_DIM / 256, C_DIM / 16, N_B);
    outproj_kernel<<<g3, 256, 0, stream>>>(x, Wo, bo, ao, out);
}
